// HighPassFilterTransformSeparately_37812892074122
// MI455X (gfx1250) — matrix operand order repaired and checked by reading, not compiled, not run on hardware
//
#include <hip/hip_runtime.h>

// CDNA5 / gfx1250: wave32 WMMA. D = A*B + C with
//   v_wmma_f32_16x16x32_bf16  (A,B bf16 16x32 / 32x16, C/D f32 16x16)
typedef __attribute__((ext_vector_type(16))) __bf16 v16bf;
typedef __attribute__((ext_vector_type(8)))  __bf16 v8bf;
typedef __attribute__((ext_vector_type(8)))  float  v8f;

#define H_IMG   1024
#define W_IMG   1024
#define TILE_Y  32           // output rows per workgroup
#define TILE_X  64           // output cols per workgroup
#define LDS_ROWS 36          // TILE_Y + 4 halo
#define LDS_W    80          // TILE_X + 16 (covers last subtile's 32-wide A read)
#define LDS_STRIDE 88        // bf16 elems/row; 176B = 11*16B (b128-aligned, bank-clean)

// ---------------------------------------------------------------------------
// Compile-time Toeplitz B table, stored per-lane as WMMA-B operand, in this kernel's contiguous k order:
//   g_BTab.v[dy][lane][i] = B_dy[K = (lane>>4)*16 + i, N = lane&15]
//   B_dy[c,n] = SRM[dy][c-n] for 0 <= c-n < 5, else 0
// All weights are small integers -> exact bf16 bit patterns, no runtime build.
// ---------------------------------------------------------------------------
struct alignas(32) BTab { unsigned short v[5][32][16]; };

constexpr unsigned short w2bf(int w) {
    return w ==   0 ? 0x0000
         : w ==  -1 ? 0xBF80      // -1.0
         : w ==   2 ? 0x4000      //  2.0
         : w ==  -2 ? 0xC000      // -2.0
         : w ==  -6 ? 0xC0C0      // -6.0
         : w ==   8 ? 0x4100      //  8.0
         :            0xC140;     // -12.0
}

constexpr BTab makeBTab() {
    BTab t{};
    const int srm[5][5] = { { -1,  2,  -2,  2, -1 },
                            {  2, -6,   8, -6,  2 },
                            { -2,  8, -12,  8, -2 },
                            {  2, -6,   8, -6,  2 },
                            { -1,  2,  -2,  2, -1 } };
    for (int dy = 0; dy < 5; ++dy)
        for (int lane = 0; lane < 32; ++lane) {
            const int n  = lane & 15;
            const int hf = lane >> 4;
            for (int i = 0; i < 16; ++i) {
                const int d = hf * 16 + i - n;      // = c - n
                const int w = (d >= 0 && d < 5) ? srm[dy][d] : 0;
                t.v[dy][lane][i] = w2bf(w);
            }
        }
    return t;
}

__constant__ BTab g_BTab = makeBTab();

__device__ __forceinline__ unsigned int pack_bf16(float a, float b) {
    const unsigned int u0 = (unsigned int)__builtin_bit_cast(unsigned short, (__bf16)a);
    const unsigned int u1 = (unsigned int)__builtin_bit_cast(unsigned short, (__bf16)b);
    return u0 | (u1 << 16);
}

__global__ __launch_bounds__(256)
void srm5_depthwise_wmma_bf16(const float* __restrict__ in,
                              float* __restrict__ out)
{
    // 36 rows * 44 dwords (88 bf16) = 6336 B of LDS
    __shared__ unsigned int lds32[LDS_ROWS * (LDS_STRIDE / 2)];
    const __bf16* ldsh = reinterpret_cast<const __bf16*>(lds32);

    const int tid  = threadIdx.x;
    const int lane = tid & 31;
    const int wave = tid >> 5;

    const int ox = blockIdx.x * TILE_X;
    const int oy = blockIdx.y * TILE_Y;
    const long long planeBase = (long long)blockIdx.z * ((long long)H_IMG * W_IMG);
    const float* __restrict__ inp  = in  + planeBase;
    float* __restrict__       outp = out + planeBase;

    // -------- load constant B fragments (coalesced, overlaps staging) --------
    v16bf bmat[5];
    #pragma unroll
    for (int dy = 0; dy < 5; ++dy) {
        const unsigned short* bp = &g_BTab.v[dy][lane][0];
        const v8bf blo = *reinterpret_cast<const v8bf*>(bp);
        const v8bf bhi = *reinterpret_cast<const v8bf*>(bp + 8);
        #pragma unroll
        for (int i = 0; i < 8; ++i) { bmat[dy][i] = blo[i]; bmat[dy][i + 8] = bhi[i]; }
    }

    // ---------------- stage fp32 -> bf16 halo tile into LDS ----------------
    // region: rows [oy-2, oy+34), cols [ox-2, ox+78); out-of-image -> 0 (conv pad)
    const bool interior = (oy >= 2) && (oy + LDS_ROWS - 2 <= H_IMG) &&
                          (ox >= 2) && (ox + LDS_W    - 2 <= W_IMG);
    if (interior) {
        // uniform fast path: unpredicated 8B-aligned float2 loads
        for (int t = tid; t < LDS_ROWS * (LDS_W / 2); t += 256) {
            const int r = t / (LDS_W / 2);
            const int p = t - r * (LDS_W / 2);
            const float2 f = *reinterpret_cast<const float2*>(
                inp + (long long)(oy - 2 + r) * W_IMG + (ox - 2 + 2 * p));
            lds32[r * (LDS_STRIDE / 2) + p] = pack_bf16(f.x, f.y);
        }
    } else {
        for (int t = tid; t < LDS_ROWS * (LDS_W / 2); t += 256) {
            const int r  = t / (LDS_W / 2);
            const int p  = t - r * (LDS_W / 2);
            const int iy = oy - 2 + r;
            const int ix = ox - 2 + 2 * p;
            float f0 = 0.0f, f1 = 0.0f;
            if ((unsigned)iy < (unsigned)H_IMG) {
                const float* rowp = inp + (long long)iy * W_IMG;
                if ((unsigned)ix       < (unsigned)W_IMG) f0 = rowp[ix];
                if ((unsigned)(ix + 1) < (unsigned)W_IMG) f1 = rowp[ix + 1];
            }
            lds32[r * (LDS_STRIDE / 2) + p] = pack_bf16(f0, f1);
        }
    }

    __syncthreads();

    // ---------------- 16x16 output subtile per wave: 5 chained WMMAs ----------------
    const int n  = lane & 15;          // A row (m) / B,D column for this lane
    const int hf = lane >> 4;
    const int sr = wave >> 2;          // 0..1  subtile row
    const int sc = wave & 3;           // 0..3  subtile col

    v8f acc;
    #pragma unroll
    for (int j = 0; j < 8; ++j) acc[j] = 0.0f;

    #pragma unroll
    for (int dy = 0; dy < 5; ++dy) {
        // A layout (16-bit, 16x32): this lane holds K run [hf*16, +16)
        const int r     = sr * 16 + n + dy;                  // LDS row
        const int colLo = sc * 16 + hf * 16;                 // 8 bf16, 16B aligned
        const int colHi = colLo + 8;
        const v8bf lo = *reinterpret_cast<const v8bf*>(ldsh + r * LDS_STRIDE + colLo);
        const v8bf hi = *reinterpret_cast<const v8bf*>(ldsh + r * LDS_STRIDE + colHi);
        v16bf a;
        #pragma unroll
        for (int i = 0; i < 8; ++i) { a[i] = lo[i]; a[i + 8] = hi[i]; }

        acc = __builtin_amdgcn_wmma_f32_16x16x32_bf16(
                  /*neg_a=*/false, a, /*neg_b=*/false, bmat[dy],
                  /*c_mod=*/(short)0, acc, /*reuse_a=*/false, /*reuse_b=*/false);
    }

    // ---------------- store D: lane n holds D[M = hf*8 + j, N = n] ----------------
    #pragma unroll
    for (int j = 0; j < 8; ++j) {
        const int Y = oy + sr * 16 + hf * 8 + j;
        const int X = ox + sc * 16 + n;
        outp[(long long)Y * W_IMG + X] = acc[j];
    }
}

extern "C" void kernel_launch(void* const* d_in, const int* in_sizes, int n_in,
                              void* d_out, int out_size, void* d_ws, size_t ws_size,
                              hipStream_t stream) {
    (void)n_in; (void)d_ws; (void)ws_size; (void)out_size;
    const float* img = (const float*)d_in[0];
    float* outp      = (float*)d_out;

    const int planes = in_sizes[0] / (H_IMG * W_IMG);   // B*C = 96
    dim3 grid(W_IMG / TILE_X, H_IMG / TILE_Y, planes);  // 16 x 32 x 96
    dim3 block(256);
    srm5_depthwise_wmma_bf16<<<grid, block, 0, stream>>>(img, outp);
}
